// SparseDenseMatMul_11879879542650
// MI455X (gfx1250) — matrix operand order repaired and checked by reading, not compiled, not run on hardware
//
#include <hip/hip_runtime.h>

// ---------------------------------------------------------------------------
// Masked batched GEMM for MI455X (gfx1250, wave32, WMMA).
// C[b,h] = (A[b,h] * mask[b]) @ B[b,h]
// A: [2,16,2048,2048] f32, mask: [2,1,2048,2048] i32, B: [2,16,2048,256] f32
// C: [2,16,2048,256] f32
//
// Roofline: 68.7 GFLOP vs 672 MB streamed -> ~29 us at 23.3 TB/s. bf16 WMMA
// (16x16x32, f32 accum) keeps compute at parity with HBM; f32 WMMA (K=4)
// would be ~8x compute-bound. Block tile covers all of N=256 so the dominant
// 512 MB A stream is read exactly once.
//
// v3: software-pipelined staging -- global loads for tile k+1 are issued
//     BEFORE the WMMAs of tile k; the loadcnt wait + bf16 convert + LDS store
//     happen after, so HBM latency hides behind matrix math.
// ---------------------------------------------------------------------------

typedef __attribute__((ext_vector_type(16))) __bf16 v16bf;
typedef __attribute__((ext_vector_type(8)))  __bf16 v8bf;
typedef __attribute__((ext_vector_type(4)))  __bf16 v4bf;
typedef __attribute__((ext_vector_type(8)))  float  v8f;
typedef __attribute__((ext_vector_type(4)))  float  v4f;
typedef __attribute__((ext_vector_type(4)))  int    v4i;

#define B_ 2
#define H_ 16
#define M_ 2048
#define K_ 2048
#define N_ 256

constexpr int BM  = 64;      // rows of C per block
constexpr int BN  = 256;     // full N, so A is streamed exactly once
constexpr int BK  = 32;      // K step (one bf16 WMMA K-depth)
constexpr int LDA = BK + 8;  // padded row stride (elems); 80B rows keep 16B align
constexpr int LDB = BK + 8;  // Bs stored N-major: [BN][LDB]

__global__ __launch_bounds__(256) void masked_gemm_bf16_wmma(
    const float* __restrict__ A,
    const int*   __restrict__ Mask,
    const float* __restrict__ Bm,
    float*       __restrict__ C)
{
    __shared__ __align__(16) __bf16 As[2][BM * LDA];   // 2 x 5.0 KB
    __shared__ __align__(16) __bf16 Bs[2][BN * LDB];   // 2 x 20.0 KB

    const int t    = threadIdx.x;
    const int lane = t & 31;
    const int wave = t >> 5;       // 0..7
    const int wm   = wave >> 2;    // 0..1 : wave row (32 rows each)
    const int wn   = wave & 3;     // 0..3 : wave col (64 cols each)
    const int ll   = lane & 15;
    const int lh   = lane >> 4;    // half-wave select

    const int bh   = blockIdx.y;          // fused (b,h) 0..31
    const int bidx = bh >> 4;             // batch index (mask has no H dim)
    const int m0   = blockIdx.x * BM;

    const float* Ab = A    + (size_t)bh   * M_ * K_;
    const int*   Mb = Mask + (size_t)bidx * M_ * K_;
    const float* Bb = Bm   + (size_t)bh   * K_ * N_;
    float*       Cb = C    + (size_t)bh   * M_ * N_;

    // Staging geometry (per thread):
    //   A tile 64x32: 2 float4 loads (coalesced along K) -> 2 ds_store_b64.
    const int ar0 = t >> 3;              // row for i=0 (0..31); i=1 adds 32
    const int ac4 = (t & 7) << 2;        // k offset 0..28
    //   B tile 32x256: 8 float4 loads (8 k-rows x 4 cols), register transpose,
    //   4 ds_store_b128 of contiguous bf16 column runs.
    const int bn4 = (t & 63) << 2;       // column base 0..252
    const int bkb = (t >> 6) << 3;       // k base 0,8,16,24

    // -- phase 1: issue global loads for tile at k0 into registers ----------
    auto stage_load = [&](int k0, v4f av[2], v4i mv[2], v4f bv[8]) {
        #pragma unroll
        for (int i = 0; i < 2; ++i) {
            const size_t g = (size_t)(m0 + ar0 + i * 32) * K_ + k0 + ac4;
            av[i] = *(const v4f*)(Ab + g);
            mv[i] = *(const v4i*)(Mb + g);
        }
        #pragma unroll
        for (int kr = 0; kr < 8; ++kr)
            bv[kr] = *(const v4f*)(Bb + (size_t)(k0 + bkb + kr) * N_ + bn4);
    };

    // -- phase 2: mask, convert to bf16, store into LDS buffer p ------------
    auto stage_store = [&](int p, const v4f av[2], const v4i mv[2],
                           const v4f bv[8]) {
        #pragma unroll
        for (int i = 0; i < 2; ++i) {
            v4bf o;
            o.x = (__bf16)(av[i].x * (float)mv[i].x);
            o.y = (__bf16)(av[i].y * (float)mv[i].y);
            o.z = (__bf16)(av[i].z * (float)mv[i].z);
            o.w = (__bf16)(av[i].w * (float)mv[i].w);
            *(v4bf*)&As[p][(ar0 + i * 32) * LDA + ac4] = o;
        }
        #pragma unroll
        for (int j = 0; j < 4; ++j) {
            v8bf col;
            #pragma unroll
            for (int kr = 0; kr < 8; ++kr)
                col[kr] = (__bf16)bv[kr][j];
            *(v8bf*)&Bs[p][(bn4 + j) * LDB + bkb] = col;
        }
    };

    v8f acc[2][4] = {};

    {   // prologue: stage tile 0 into buffer 0
        v4f av[2]; v4i mv[2]; v4f bv[8];
        stage_load(0, av, mv, bv);
        stage_store(0, av, mv, bv);
    }
    int p = 0;

    for (int k0 = 0; k0 < K_; k0 += BK) {
        __syncthreads();                   // buffer p ready; buffer p^1 free

        const bool nxt = (k0 + BK < K_);
        v4f av[2]; v4i mv[2]; v4f bv[8];
        if (nxt)
            stage_load(k0 + BK, av, mv, bv);   // issue loads, no wait yet

        // ---- compute: 8 x v_wmma_f32_16x16x32_bf16 per wave ----------------
        // A frag (16-bit A 16x32): lane (m=ll, half=lh) holds
        //   K = lh*8 + {0..7} in VGPR0-3 and K = lh*8 + {16..23} in VGPR4-7.
        v16bf afrag[2];
        #pragma unroll
        for (int fm = 0; fm < 2; ++fm) {
            const int row = wm * 32 + fm * 16 + ll;
            const int kb  = lh * 8;
            union { v16bf v; v8bf h[2]; } u;
            u.h[0] = *(const v8bf*)&As[p][row * LDA + kb];
            u.h[1] = *(const v8bf*)&As[p][row * LDA + kb + 16];
            afrag[fm] = u.v;
        }
        // B frag (16-bit B 32x16): lane (n=ll, half=lh) holds column n,
        //   K = lh*8 + {0..7} and K = lh*8 + {16..23} (matches A frag).
        #pragma unroll
        for (int fn = 0; fn < 4; ++fn) {
            const int col = wn * 64 + fn * 16 + ll;
            const int kb  = lh * 8;
            union { v16bf v; v8bf h[2]; } u;
            u.h[0] = *(const v8bf*)&Bs[p][col * LDB + kb];
            u.h[1] = *(const v8bf*)&Bs[p][col * LDB + kb + 16];
            const v16bf bfrag = u.v;
            #pragma unroll
            for (int fm = 0; fm < 2; ++fm) {
                acc[fm][fn] = __builtin_amdgcn_wmma_f32_16x16x32_bf16(
                    /*neg_a=*/false, afrag[fm],
                    /*neg_b=*/false, bfrag,
                    /*c_mod=*/(short)0, acc[fm][fn],
                    /*reuse_a=*/false, /*reuse_b=*/false);
            }
        }

        // ---- drain staged registers into the other LDS buffer --------------
        if (nxt)
            stage_store(p ^ 1, av, mv, bv);    // loadcnt wait lands here

        p ^= 1;
    }

    // ---- epilogue: C/D layout: VGPR r, lanes 0-15 -> M=r, lanes 16-31 -> M=r+8
    #pragma unroll
    for (int fm = 0; fm < 2; ++fm) {
        #pragma unroll
        for (int fn = 0; fn < 4; ++fn) {
            const int col = wn * 64 + fn * 16 + ll;
            #pragma unroll
            for (int r = 0; r < 8; ++r) {
                const int row = m0 + wm * 32 + fm * 16 + r + lh * 8;
                Cb[(size_t)row * N_ + col] = acc[fm][fn][r];
            }
        }
    }
}

extern "C" void kernel_launch(void* const* d_in, const int* in_sizes, int n_in,
                              void* d_out, int out_size, void* d_ws, size_t ws_size,
                              hipStream_t stream) {
    const float* a    = (const float*)d_in[0];
    const int*   mask = (const int*)d_in[1];
    const float* b    = (const float*)d_in[2];
    float*       out  = (float*)d_out;

    dim3 grid(M_ / BM, B_ * H_);   // 32 x 32 blocks
    dim3 block(256);               // 8 waves (wave32)
    masked_gemm_bf16_wmma<<<grid, block, 0, stream>>>(a, mask, b, out);
}
